// Attention_66803921322307
// MI455X (gfx1250) — matrix operand order repaired and checked by reading, not compiled, not run on hardware
//
#include <hip/hip_runtime.h>
#include <hip/hip_bf16.h>

// ---------------- types ----------------
typedef __bf16 bf16_t;
typedef bf16_t v16bf __attribute__((ext_vector_type(16)));
typedef bf16_t v8bf  __attribute__((ext_vector_type(8)));
typedef float  v8f   __attribute__((ext_vector_type(8)));
typedef unsigned int u32x4 __attribute__((ext_vector_type(4)));
typedef int i32x4 __attribute__((ext_vector_type(4)));
typedef int i32x8 __attribute__((ext_vector_type(8)));

#define S_LEN 2048
#define D_MODEL 4096
#define NH 32
#define NKV 8
#define HD 128
#define N_REP 4

#if __has_builtin(__builtin_amdgcn_tensor_load_to_lds)
#define HAVE_TDM 1
#else
#define HAVE_TDM 0
#endif

__device__ __forceinline__ unsigned short f2bf(float x) {
    unsigned int u = __float_as_uint(x);
    unsigned int r = u + 0x7FFFu + ((u >> 16) & 1u);
    return (unsigned short)(r >> 16);
}

// load 16 bf16: 8 at p, 8 at p+hi_off (elements); both 16B aligned
__device__ __forceinline__ v16bf load_frag(const unsigned short* p, int hi_off) {
    v8bf lo = *(const v8bf*)p;
    v8bf hi = *(const v8bf*)(p + hi_off);
    return __builtin_shufflevector(lo, hi, 0,1,2,3,4,5,6,7,8,9,10,11,12,13,14,15);
}

__device__ __forceinline__ v8f wmma_bf16(v16bf a, v16bf b, v8f c) {
    return __builtin_amdgcn_wmma_f32_16x16x32_bf16(false, a, false, b, (short)0, c, false, false);
}

__device__ __forceinline__ float hmax16(float t) {
    t = fmaxf(t, __shfl_xor(t, 1));
    t = fmaxf(t, __shfl_xor(t, 2));
    t = fmaxf(t, __shfl_xor(t, 4));
    t = fmaxf(t, __shfl_xor(t, 8));
    return t;
}
__device__ __forceinline__ float hsum16(float t) {
    t += __shfl_xor(t, 1);
    t += __shfl_xor(t, 2);
    t += __shfl_xor(t, 4);
    t += __shfl_xor(t, 8);
    return t;
}

#if HAVE_TDM
// Issue a 2D TDM tile load (bf16 elements) per CDNA5 D# layout (ISA 08 §8.3-8.5).
// tensor_d0/d1: tensor extents (elements); stride0: dim0 row stride (elements);
// tile_d0 x tile_d1: tile shape (dim0 contiguous).
__device__ __forceinline__ void tdm_load_2d(unsigned int lds_off, const void* gaddr,
                                            unsigned int tensor_d0, unsigned int tensor_d1,
                                            unsigned long long stride0,
                                            unsigned int tile_d0, unsigned int tile_d1) {
    const unsigned long long ga = (unsigned long long)gaddr;
    u32x4 g0;
    g0[0] = 1u;                                   // count=1, user descriptor
    g0[1] = lds_off;                              // lds_addr [63:32]
    g0[2] = (unsigned int)ga;                     // global_addr [95:64]
    g0[3] = (unsigned int)((ga >> 32) & 0x1FFFFFFull) | (2u << 30);  // ga[56:32], type=2
    i32x8 g1;
    g1[0] = (int)(1u << 16);                      // wg_mask=0, data_size=1 (2B)
    g1[1] = (int)((tensor_d0 & 0xFFFFu) << 16);   // atomic_addr=0 | tensor_dim0[15:0]
    g1[2] = (int)((tensor_d0 >> 16) | ((tensor_d1 & 0xFFFFu) << 16));
    g1[3] = (int)((tensor_d1 >> 16) | (tile_d0 << 16));              // | tile_dim0
    g1[4] = (int)(tile_d1 & 0xFFFFu);             // tile_dim1, tile_dim2=0
    g1[5] = (int)(unsigned int)(stride0 & 0xFFFFFFFFull);            // dim0 stride lo
    g1[6] = (int)(unsigned int)((stride0 >> 32) & 0xFFFFull);        // dim0 stride hi
    g1[7] = 0;                                    // dim1 stride (unused, tile_dim2=0)
    i32x4 z4 = (i32x4){0, 0, 0, 0};
#if __clang_major__ >= 23
    i32x8 z8 = (i32x8){0, 0, 0, 0, 0, 0, 0, 0};
    __builtin_amdgcn_tensor_load_to_lds(g0, g1, z4, z4, z8, 0);
#else
    __builtin_amdgcn_tensor_load_to_lds(g0, g1, z4, z4, 0);
#endif
}
#endif

// ---------------- GEMM: C[M,N] = A[M,K] * W[K,N], fp32 in/out, bf16 WMMA ----------------
#define TILE_M 128
#define TILE_N 128
#define TILE_K 32
#define LDSK 48   // padded row stride (bf16 elems); 96B keeps 16B alignment

__global__ __launch_bounds__(256)
void gemm_bf16wmma(const float* __restrict__ A, const float* __restrict__ W,
                   float* __restrict__ C, int M, int N, int K) {
    __shared__ unsigned short sA[TILE_M * LDSK];  // [row][k]
    __shared__ unsigned short sB[TILE_N * LDSK];  // [n][k]  (W transposed)
    const int tid  = threadIdx.x;
    const int wave = tid >> 5, lane = tid & 31;
    const int lr = lane & 15, hv = lane >> 4;
    const int m0 = blockIdx.y * TILE_M, n0 = blockIdx.x * TILE_N;
    const int wm = (wave & 3) * 32;   // 4 waves stacked in M
    const int wn = (wave >> 2) * 64;  // 2 waves in N

    v8f acc[2][4];
#pragma unroll
    for (int mi = 0; mi < 2; ++mi)
#pragma unroll
        for (int ni = 0; ni < 4; ++ni) acc[mi][ni] = (v8f){};

    const int kqA = (tid & 7) * 4;   // k offset for A staging
    const int rbA = tid >> 3;        // 0..31
    const int nqB = (tid & 31) * 4;  // n offset for W staging
    const int kbB = tid >> 5;        // 0..7

    for (int k0 = 0; k0 < K; k0 += TILE_K) {
        // stage A (fp32 -> bf16), row-major
#pragma unroll
        for (int ro = 0; ro < TILE_M; ro += 32) {
            const float4 f = *(const float4*)(A + (size_t)(m0 + rbA + ro) * K + k0 + kqA);
            unsigned short* p = &sA[(rbA + ro) * LDSK + kqA];
            p[0] = f2bf(f.x); p[1] = f2bf(f.y); p[2] = f2bf(f.z); p[3] = f2bf(f.w);
        }
        // stage W transposed (fp32 -> bf16): sB[n][k]
#pragma unroll
        for (int ko = 0; ko < TILE_K; ko += 8) {
            const int kk = kbB + ko;
            const float4 f = *(const float4*)(W + (size_t)(k0 + kk) * N + n0 + nqB);
            sB[(nqB + 0) * LDSK + kk] = f2bf(f.x);
            sB[(nqB + 1) * LDSK + kk] = f2bf(f.y);
            sB[(nqB + 2) * LDSK + kk] = f2bf(f.z);
            sB[(nqB + 3) * LDSK + kk] = f2bf(f.w);
        }
        if (k0 + TILE_K < K) {  // hint next slabs into cache
            __builtin_prefetch(A + (size_t)(m0 + rbA) * K + k0 + TILE_K + kqA, 0, 1);
            __builtin_prefetch(W + (size_t)(k0 + TILE_K + kbB) * N + n0 + nqB, 0, 1);
        }
        __syncthreads();

        v16bf afrag[2], bfrag[4];
#pragma unroll
        for (int mi = 0; mi < 2; ++mi) {
            const int row = wm + mi * 16 + lr;
            afrag[mi] = load_frag(&sA[row * LDSK + 8 * hv], 16);   // K {0..7,16..23} / {8..15,24..31}
        }
#pragma unroll
        for (int ni = 0; ni < 4; ++ni) {
            const int col = wn + ni * 16 + lr;
            bfrag[ni] = load_frag(&sB[col * LDSK + 8 * hv], 16);   // K {0..7,16..23} / {8..15,24..31}
        }
#pragma unroll
        for (int mi = 0; mi < 2; ++mi)
#pragma unroll
            for (int ni = 0; ni < 4; ++ni)
                acc[mi][ni] = wmma_bf16(afrag[mi], bfrag[ni], acc[mi][ni]);
        __syncthreads();
    }

    // C layout: lane lr = n, VGPR r -> m = r + 8*hv
#pragma unroll
    for (int mi = 0; mi < 2; ++mi)
#pragma unroll
        for (int ni = 0; ni < 4; ++ni)
#pragma unroll
            for (int r = 0; r < 8; ++r) {
                const int m = m0 + wm + mi * 16 + r + 8 * hv;
                const int n = n0 + wn + ni * 16 + lr;
                C[(size_t)m * N + n] = acc[mi][ni][r];
            }
}

// ---------------- RoPE + bf16, [S][H][HD] fp32 -> [H][S][HD] bf16 ----------------
__global__ void rope_to_bf16(const float* __restrict__ src, const float* __restrict__ cosd,
                             const float* __restrict__ sind, unsigned short* __restrict__ dst,
                             int nheads, int total) {
    int idx = blockIdx.x * blockDim.x + threadIdx.x;
    if (idx >= total) return;
    const int i = idx & 63;
    const int h = (idx >> 6) % nheads;
    const int s = idx / (64 * nheads);
    const size_t si = ((size_t)s * nheads + h) * HD + 2 * i;
    const float tr = src[si], ti = src[si + 1];
    const float c = cosd[s * 64 + i], sn = sind[s * 64 + i];
    const size_t o = ((size_t)h * S_LEN + s) * HD + 2 * i;
    dst[o]     = f2bf(tr * c - ti * sn);
    dst[o + 1] = f2bf(tr * sn + ti * c);
}

// ---------------- V: [S][NKV][HD] fp32 -> [NKV][HD][S] bf16 (transposed) ----------------
__global__ void v_transpose_bf16(const float* __restrict__ src, unsigned short* __restrict__ dst,
                                 int total) {
    int idx = blockIdx.x * blockDim.x + threadIdx.x;
    if (idx >= total) return;
    const int d = idx & 127;
    const int h = (idx >> 7) % NKV;
    const int s = idx / (HD * NKV);
    dst[((size_t)h * HD + d) * S_LEN + s] = f2bf(src[((size_t)s * NKV + h) * HD + d]);
}

// ---------------- causal flash attention, bf16 WMMA, TDM-staged K/V tiles ----------------
__global__ __launch_bounds__(128)
void flash_attn(const unsigned short* __restrict__ Qbf, const unsigned short* __restrict__ Kbf,
                const unsigned short* __restrict__ Vt, float* __restrict__ AO) {
    __shared__ unsigned short sK[32 * HD];        // [key][d]   8 KB
    __shared__ unsigned short sV[HD * 32];        // [d][key]   8 KB
    __shared__ unsigned short sP[4][16 * LDSK];   // per-wave P tile (16x32, padded)
    const int tid = threadIdx.x, wave = tid >> 5, lane = tid & 31;
    const int lr = lane & 15, hv = lane >> 4;
    const int h = blockIdx.y, hkv = h >> 2;       // N_REP = 4
    const int qb = blockIdx.x * 64 + wave * 16;   // this wave's 16 query rows

    const unsigned short* Qh = Qbf + (size_t)h   * S_LEN * HD;
    const unsigned short* Kh = Kbf + (size_t)hkv * S_LEN * HD;
    const unsigned short* Vh = Vt  + (size_t)hkv * HD * S_LEN;

    // Q fragments, reused for every key block (A layout: lane = row, K split by half)
    v16bf qf[4];
#pragma unroll
    for (int kk = 0; kk < 4; ++kk)
        qf[kk] = load_frag(Qh + (size_t)(qb + lr) * HD + kk * 32 + 8 * hv, 16);

    float mrow[8], lrow[8];
    v8f oacc[8];
#pragma unroll
    for (int r = 0; r < 8; ++r) { mrow[r] = -1e30f; lrow[r] = 0.0f; }
#pragma unroll
    for (int n = 0; n < 8; ++n) oacc[n] = (v8f){};

    unsigned short* P = sP[wave];
    const float scale = 0.08838834764831845f;        // 1/sqrt(128)
    const int kb_end  = (qb + 47) >> 5;              // this wave's causal bound
    const int kb_blk  = (blockIdx.x * 64 + 95) >> 5; // block max (wave 3's bound)

#if HAVE_TDM
    const unsigned int ldsK_off = (unsigned int)(size_t)(void*)sK;
    const unsigned int ldsV_off = (unsigned int)(size_t)(void*)sV;
#endif

    for (int kb = 0; kb < kb_blk; ++kb) {
        const int key0 = kb * 32;
        __syncthreads();   // previous tile fully consumed
#if HAVE_TDM
        if (wave == 0) {
            // K tile: 32 keys x 128 d, row-contiguous in Kbf
            tdm_load_2d(ldsK_off, Kh + (size_t)key0 * HD, HD, S_LEN, HD, HD, 32);
            // V tile: 128 d rows x 32 keys, rows strided by S_LEN in Vt
            tdm_load_2d(ldsV_off, Vh + (size_t)key0, S_LEN, HD, S_LEN, 32, HD);
            __builtin_amdgcn_s_wait_tensorcnt(0);
        }
#else
        {   // cooperative fallback staging
            uint4* dK = (uint4*)sK;
            const uint4* gK = (const uint4*)(Kh + (size_t)key0 * HD);
            for (int i = tid; i < 512; i += 128) dK[i] = gK[i];
            uint4* dV = (uint4*)sV;
            for (int i = tid; i < 512; i += 128) {
                const int d = i >> 2, part = i & 3;
                dV[i] = *(const uint4*)(Vh + (size_t)d * S_LEN + key0 + part * 8);
            }
        }
#endif
        __syncthreads();   // tile visible to all waves

        if (kb < kb_end) {
            v8f s0 = (v8f){}, s1 = (v8f){};
#pragma unroll
            for (int kk = 0; kk < 4; ++kk) {  // chain over HD=128
                v16bf b0 = load_frag(&sK[(     lr) * HD + kk * 32 + 8 * hv], 16);
                v16bf b1 = load_frag(&sK[(16 + lr) * HD + kk * 32 + 8 * hv], 16);
                s0 = wmma_bf16(qf[kk], b0, s0);
                s1 = wmma_bf16(qf[kk], b1, s1);
            }
            // scale + causal mask + online softmax
#pragma unroll
            for (int r = 0; r < 8; ++r) {
                const int qrow = qb + r + 8 * hv;
                float a = s0[r] * scale; if (key0 +      lr > qrow) a = -1e30f;
                float b = s1[r] * scale; if (key0 + 16 + lr > qrow) b = -1e30f;
                const float rmax = hmax16(fmaxf(a, b));
                const float mn = fmaxf(mrow[r], rmax);
                const float alpha = __expf(mrow[r] - mn);
                const float p0 = __expf(a - mn);
                const float p1 = __expf(b - mn);
                lrow[r] = lrow[r] * alpha + hsum16(p0 + p1);
                mrow[r] = mn;
#pragma unroll
                for (int n = 0; n < 8; ++n) oacc[n][r] *= alpha;
                P[(r + 8 * hv) * LDSK +      lr] = f2bf(p0);
                P[(r + 8 * hv) * LDSK + 16 + lr] = f2bf(p1);
            }
            // P as A operand (same-wave DS ops are in-order)
            const v16bf pa = load_frag(P + lr * LDSK + 8 * hv, 16);
            // P(16x32) x V(32xHD): sV gives contiguous keys per d-column
#pragma unroll
            for (int n = 0; n < 8; ++n) {
                v16bf vb = load_frag(&sV[(n * 16 + lr) * 32 + 8 * hv], 16);
                oacc[n] = wmma_bf16(pa, vb, oacc[n]);
            }
        }
    }

#pragma unroll
    for (int r = 0; r < 8; ++r) {
        const float inv = 1.0f / lrow[r];
        const size_t row = (size_t)(qb + r + 8 * hv) * (NH * HD) + (size_t)h * HD;
#pragma unroll
        for (int n = 0; n < 8; ++n)
            AO[row + n * 16 + lr] = oacc[n][r] * inv;
    }
}

// ---------------- launch ----------------
extern "C" void kernel_launch(void* const* d_in, const int* in_sizes, int n_in,
                              void* d_out, int out_size, void* d_ws, size_t ws_size,
                              hipStream_t stream) {
    (void)in_sizes; (void)n_in; (void)out_size; (void)ws_size;
    const float* x   = (const float*)d_in[0];
    const float* wq  = (const float*)d_in[1];
    const float* wk  = (const float*)d_in[2];
    const float* wv  = (const float*)d_in[3];
    const float* wo  = (const float*)d_in[4];
    const float* cosd = (const float*)d_in[5];
    const float* sind = (const float*)d_in[6];
    float* out = (float*)d_out;

    char* ws = (char*)d_ws;
    float*          q32 = (float*)(ws);                         // 32 MB
    float*          k32 = (float*)(ws + 33554432);              //  8 MB
    float*          v32 = (float*)(ws + 41943040);              //  8 MB
    unsigned short* qbf = (unsigned short*)(ws + 50331648);     // 16 MB
    unsigned short* kbf = (unsigned short*)(ws + 67108864);     //  4 MB
    unsigned short* vt  = (unsigned short*)(ws + 71303168);     //  4 MB
    float*          ao  = (float*)(ws + 75497472);              // 32 MB

    // QKV projections
    gemm_bf16wmma<<<dim3(D_MODEL / TILE_N, S_LEN / TILE_M), 256, 0, stream>>>(x, wq, q32, S_LEN, D_MODEL, D_MODEL);
    gemm_bf16wmma<<<dim3((NKV * HD) / TILE_N, S_LEN / TILE_M), 256, 0, stream>>>(x, wk, k32, S_LEN, NKV * HD, D_MODEL);
    gemm_bf16wmma<<<dim3((NKV * HD) / TILE_N, S_LEN / TILE_M), 256, 0, stream>>>(x, wv, v32, S_LEN, NKV * HD, D_MODEL);

    // RoPE + bf16 pack (head-major), V transpose
    {
        int tq = S_LEN * NH * (HD / 2);
        rope_to_bf16<<<(tq + 255) / 256, 256, 0, stream>>>(q32, cosd, sind, qbf, NH, tq);
        int tk = S_LEN * NKV * (HD / 2);
        rope_to_bf16<<<(tk + 255) / 256, 256, 0, stream>>>(k32, cosd, sind, kbf, NKV, tk);
        int tv = S_LEN * NKV * HD;
        v_transpose_bf16<<<(tv + 255) / 256, 256, 0, stream>>>(v32, vt, tv);
    }

    // causal flash attention (TDM-staged K/V tiles)
    flash_attn<<<dim3(S_LEN / 64, NH), 128, 0, stream>>>(qbf, kbf, vt, ao);

    // output projection
    gemm_bf16wmma<<<dim3(D_MODEL / TILE_N, S_LEN / TILE_M), 256, 0, stream>>>(ao, wo, out, S_LEN, D_MODEL, NH * HD);
}
